// TransformerEncoder_65317862637761
// MI455X (gfx1250) — matrix operand order repaired and checked by reading, not compiled, not run on hardware
//
#include <hip/hip_runtime.h>

// ---------------------------------------------------------------------------
// Types / WMMA helpers (CDNA5 / gfx1250, wave32)
// ---------------------------------------------------------------------------
typedef _Float16 h8   __attribute__((ext_vector_type(8)));
typedef _Float16 h16v __attribute__((ext_vector_type(16)));
typedef float    f8   __attribute__((ext_vector_type(8)));

__device__ __forceinline__ h16v mk16(const _Float16* p0, const _Float16* p1) {
  h8 lo = *(const h8*)p0;
  h8 hi = *(const h8*)p1;
  return __builtin_shufflevector(lo, hi, 0,1,2,3,4,5,6,7,8,9,10,11,12,13,14,15);
}

__device__ __forceinline__ f8 wmma_f16(h16v a, h16v b, f8 c) {
  // D(16x16,f32) = A(16x32,f16) * B(32x16,f16) + C
  return __builtin_amdgcn_wmma_f32_16x16x32_f16(false, a, false, b, (short)0, c,
                                                false, false);
}

// Fragment layouts (per CDNA5 ISA 7.12.2, wave32):
//  A 16x32 f16 : lane<16 -> row=lane,  elems[0..7]=K0..7,  elems[8..15]=K16..23
//                lane>=16-> row=lane-16,elems[0..7]=K8..15, elems[8..15]=K24..31
//     => per lane: two contiguous half8 loads at kA=(lane<16?0:8) and kA+16
//  B 32x16 f16 : lane<16 -> col=lane,   elems[0..7]=K0..7,  elems[8..15]=K16..23
//                lane>=16-> col=lane-16,elems[0..7]=K8..15, elems[8..15]=K24..31
//     => per lane: two half8 loads from Bt[col][kB..] and Bt[col][kB+16..], kB=(lane<16?0:8)
//  C/D 16x16 f32: vgpr r -> row = r + 8*(lane>>4), col = lane&15

// ---------------------------------------------------------------------------
// Constants
// ---------------------------------------------------------------------------
#define NTOK 2048   // B * (S0+1) = 8 * 256
#define DM   512
#define DKV  64
#define NH   8
#define FH   128
#define NL   4

// ---------------------------------------------------------------------------
// h0 = concat(x, event_token) ; writes f32 + f16
// ---------------------------------------------------------------------------
__global__ void k_prep_h0(const float* __restrict__ x, const float* __restrict__ ev,
                          float* __restrict__ h, _Float16* __restrict__ hf) {
  int i = blockIdx.x * blockDim.x + threadIdx.x;
  if (i >= NTOK * DM) return;
  int d = i & (DM - 1), n = i >> 9;
  int s = n & 255, b = n >> 8;
  float v = (s < 255) ? x[((size_t)(b * 255 + s) << 9) + d] : ev[d];
  h[i] = v;
  hf[i] = (_Float16)v;
}

// ---------------------------------------------------------------------------
// Weight prep: W[K][Nc] f32 -> Wt[Nc][K] f16 (K contiguous => B-fragment friendly)
// ---------------------------------------------------------------------------
__global__ void k_prep_wt(const float* __restrict__ w, _Float16* __restrict__ wt,
                          int K, int Nc) {
  int i = blockIdx.x * blockDim.x + threadIdx.x;
  if (i >= K * Nc) return;
  int n = i / K, k = i - n * K;
  wt[i] = (_Float16)w[(size_t)k * Nc + n];
}

// V[h][n][d] -> Vt[h][d][n]  (token axis contiguous for B fragments of P@V)
__global__ void k_vt(const _Float16* __restrict__ v, _Float16* __restrict__ vt) {
  int i = blockIdx.x * blockDim.x + threadIdx.x;
  if (i >= NH * NTOK * DKV) return;
  int d = i & 63;
  int n = (i >> 6) & (NTOK - 1);
  int hh = i >> 17;
  vt[(((size_t)hh * DKV + d) << 11) + n] = v[i];
}

// ---------------------------------------------------------------------------
// Batched WMMA GEMM: for each batch b:
//   C[b][M][Nc] = A[M][K](f16) * Bt[b][Nc][K](f16)^T + bias[b], opt ReLU
// One wave per 16x32 output tile pair (A fragment reused across 2 WMMAs/k-step).
// K % 32 == 0, M % 16 == 0, Nc % 32 == 0. A is shared across the batch.
// ---------------------------------------------------------------------------
__global__ void __launch_bounds__(256)
k_gemm(const _Float16* __restrict__ A, const _Float16* __restrict__ Bt,
       const float* __restrict__ bias, float* __restrict__ Cf,
       _Float16* __restrict__ Ch, int M, int K, int Nc, int relu,
       int batch, long sB, long sBias, long sC) {
  const int lane = threadIdx.x & 31;
  const int wv   = threadIdx.x >> 5;
  const int nPair = Nc >> 5;                      // pairs of 16-col tiles
  const int per  = (M >> 4) * nPair;
  const int job  = blockIdx.x * 8 + wv;
  if (job >= per * batch) return;                 // wave-uniform: EXEC stays all-1
  const int bi = job / per;
  const int jp = job - bi * per;
  const int tM = jp / nPair, tP = jp - tM * nPair;
  const int l16 = lane & 15, hi = lane >> 4;
  const int kA = hi ? 8 : 0, kB = hi ? 8 : 0;
  const int row = (tM << 4) + l16;
  const int col = (tP << 5) + l16;
  const _Float16* ap  = A + (size_t)row * K;
  const _Float16* bp0 = Bt + (size_t)bi * sB + (size_t)col * K;
  const _Float16* bp1 = bp0 + (size_t)16 * K;
  const float*   bpb  = bias + (size_t)bi * sBias;
  f8 acc0 = {0.f, 0.f, 0.f, 0.f, 0.f, 0.f, 0.f, 0.f};
  f8 acc1 = acc0;
  for (int k0 = 0; k0 < K; k0 += 32) {
    // speculative prefetch two k-steps ahead, WGP scope (pull into near caches)
    __builtin_prefetch(ap + k0 + 64, 0, 3);
    h16v a  = mk16(ap + k0 + kA, ap + k0 + kA + 16);
    h16v b0 = mk16(bp0 + k0 + kB, bp0 + k0 + kB + 16);
    h16v b1 = mk16(bp1 + k0 + kB, bp1 + k0 + kB + 16);
    acc0 = wmma_f16(a, b0, acc0);
    acc1 = wmma_f16(a, b1, acc1);
  }
  const float bb0 = bpb[col];
  const float bb1 = bpb[col + 16];
#pragma unroll
  for (int r = 0; r < 8; ++r) {
    float v0 = acc0[r] + bb0;
    float v1 = acc1[r] + bb1;
    if (relu) { v0 = fmaxf(v0, 0.f); v1 = fmaxf(v1, 0.f); }
    const int orow = (tM << 4) + r + (hi << 3);
    const size_t oi = (size_t)bi * sC + (size_t)orow * Nc + col;
    if (Cf) { Cf[oi] = v0; Cf[oi + 16] = v1; }
    if (Ch) { Ch[oi] = (_Float16)v0; Ch[oi + 16] = (_Float16)v1; }
  }
}

// ---------------------------------------------------------------------------
// Flash attention over flattened 2048-token axis.
// One wave per (head, 16-row tile). Streams 32-column blocks.
// Q,K: [H][N][64] f16 row-major ; Vt: [H][64][N] f16 ; out: concat[N][H*64] f16
// ---------------------------------------------------------------------------
__global__ void __launch_bounds__(256)
k_flash(const _Float16* __restrict__ Q, const _Float16* __restrict__ Kh,
        const _Float16* __restrict__ Vt, _Float16* __restrict__ concat) {
  __shared__ _Float16 plds[8][16 * 32];           // per-wave 16x32 P tile (1 KB)
  const int lane = threadIdx.x & 31;
  const int wv   = threadIdx.x >> 5;
  const int job  = blockIdx.x * 8 + wv;
  if (job >= NH * (NTOK / 16)) return;
  const int hh = job >> 7;                        // head
  const int rt = job & 127;                       // 16-row tile index
  const int l16 = lane & 15, hi = lane >> 4;
  const int kA = hi ? 8 : 0, kB = hi ? 8 : 0;
  const int rowbase = rt << 4;

  // Q row fragments for K=0..63 (2 k-steps), kept in registers
  const _Float16* qp = Q + (((size_t)hh << 11) + rowbase + l16) * DKV;
  const h16v a0 = mk16(qp + kA, qp + kA + 16);
  const h16v a1 = mk16(qp + 32 + kA, qp + 32 + kA + 16);

  f8 acc[4];
  const f8 z8 = {0.f, 0.f, 0.f, 0.f, 0.f, 0.f, 0.f, 0.f};
#pragma unroll
  for (int t = 0; t < 4; ++t) acc[t] = z8;
  float rmax[8], rsum[8];
#pragma unroll
  for (int r = 0; r < 8; ++r) { rmax[r] = -3.0e38f; rsum[r] = 0.f; }

  _Float16* pl = plds[wv];

  for (int j = 0; j < NTOK / 32; ++j) {
    const int cb = j << 5;
    // ---- scores: S(16x32) = Q(16x64) @ K^T, two 16-col tiles, 2 k-steps each
    f8 s0 = z8, s1 = z8;
    {
      const _Float16* kp = Kh + (((size_t)hh << 11) + cb + l16) * DKV;
      h16v b0 = mk16(kp + kB, kp + kB + 16);
      h16v b1 = mk16(kp + 32 + kB, kp + 32 + kB + 16);
      s0 = wmma_f16(a0, b0, s0);
      s0 = wmma_f16(a1, b1, s0);
      kp += (size_t)16 * DKV;
      b0 = mk16(kp + kB, kp + kB + 16);
      b1 = mk16(kp + 32 + kB, kp + 32 + kB + 16);
      s1 = wmma_f16(a0, b0, s1);
      s1 = wmma_f16(a1, b1, s1);
    }
    // ---- online softmax (row stats across 16 lanes holding each row)
    float p0[8], p1[8], fac[8];
#pragma unroll
    for (int r = 0; r < 8; ++r) {
      const float v0 = (float)s0[r] * 0.125f;     // 1/sqrt(64)
      const float v1 = (float)s1[r] * 0.125f;
      float mx = fmaxf(v0, v1);
      mx = fmaxf(mx, __shfl_xor(mx, 1, 32));
      mx = fmaxf(mx, __shfl_xor(mx, 2, 32));
      mx = fmaxf(mx, __shfl_xor(mx, 4, 32));
      mx = fmaxf(mx, __shfl_xor(mx, 8, 32));
      const float nm = fmaxf(rmax[r], mx);
      const float f = __expf(rmax[r] - nm);
      const float e0 = __expf(v0 - nm);
      const float e1 = __expf(v1 - nm);
      float rs = e0 + e1;
      rs += __shfl_xor(rs, 1, 32);
      rs += __shfl_xor(rs, 2, 32);
      rs += __shfl_xor(rs, 4, 32);
      rs += __shfl_xor(rs, 8, 32);
      rsum[r] = rsum[r] * f + rs;
      rmax[r] = nm;
      fac[r] = f; p0[r] = e0; p1[r] = e1;
    }
#pragma unroll
    for (int r = 0; r < 8; ++r) {
      acc[0][r] *= fac[r]; acc[1][r] *= fac[r];
      acc[2][r] *= fac[r]; acc[3][r] *= fac[r];
    }
    // ---- P tile (C layout) -> LDS -> A-fragment layout
#pragma unroll
    for (int r = 0; r < 8; ++r) {
      const int rr = r + (hi << 3);
      pl[rr * 32 + l16]      = (_Float16)p0[r];
      pl[rr * 32 + 16 + l16] = (_Float16)p1[r];
    }
    asm volatile("s_wait_dscnt 0" ::: "memory");   // cross-lane LDS visibility (intra-wave)
    const h16v pf = mk16(pl + l16 * 32 + kA, pl + l16 * 32 + kA + 16);
    // ---- O(16x64) += P(16x32) @ V(32x64) : 4 column tiles from Vt
#pragma unroll
    for (int t = 0; t < 4; ++t) {
      const _Float16* vp =
          Vt + ((((size_t)hh << 6) + (t << 4) + l16) << 11) + cb + kB;
      const h16v bv = mk16(vp, vp + 16);
      acc[t] = wmma_f16(pf, bv, acc[t]);
    }
    asm volatile("s_wait_dscnt 0" ::: "memory");   // reads done before next overwrite
  }
  // ---- normalize + head-major concat store
#pragma unroll
  for (int r = 0; r < 8; ++r) {
    const float inv = 1.0f / rsum[r];
    const int orow = rowbase + r + (hi << 3);
#pragma unroll
    for (int t = 0; t < 4; ++t) {
      concat[(size_t)orow * DM + (hh << 6) + (t << 4) + l16] =
          (_Float16)(acc[t][r] * inv);
    }
  }
}

// ---------------------------------------------------------------------------
// AddNorm: out = LN(a + b) * g + beta ; writes f32 + f16. One wave per row.
// ---------------------------------------------------------------------------
__global__ void __launch_bounds__(256)
k_addnorm(const float* __restrict__ a, const float* __restrict__ b,
          const float* __restrict__ g, const float* __restrict__ be,
          float* __restrict__ out, _Float16* __restrict__ outh, int rows) {
  const int lane = threadIdx.x & 31;
  const int wv   = threadIdx.x >> 5;
  const int row = blockIdx.x * 8 + wv;
  if (row >= rows) return;
  const size_t base = (size_t)row << 9;
  float zv[16];
  float s = 0.f;
#pragma unroll
  for (int i = 0; i < 16; ++i) {
    const int d = lane + (i << 5);
    zv[i] = a[base + d] + b[base + d];
    s += zv[i];
  }
  s += __shfl_xor(s, 1, 32);  s += __shfl_xor(s, 2, 32);
  s += __shfl_xor(s, 4, 32);  s += __shfl_xor(s, 8, 32);
  s += __shfl_xor(s, 16, 32);
  const float mean = s * (1.0f / 512.0f);
  float vs = 0.f;
#pragma unroll
  for (int i = 0; i < 16; ++i) { const float d0 = zv[i] - mean; vs += d0 * d0; }
  vs += __shfl_xor(vs, 1, 32);  vs += __shfl_xor(vs, 2, 32);
  vs += __shfl_xor(vs, 4, 32);  vs += __shfl_xor(vs, 8, 32);
  vs += __shfl_xor(vs, 16, 32);
  const float rstd = rsqrtf(vs * (1.0f / 512.0f) + 1e-5f);
#pragma unroll
  for (int i = 0; i < 16; ++i) {
    const int d = lane + (i << 5);
    const float o = (zv[i] - mean) * rstd * g[d] + be[d];
    out[base + d] = o;
    outh[base + d] = (_Float16)o;
  }
}

// out[b][d] = h[b*256 + 255][d]
__global__ void k_extract(const float* __restrict__ h, float* __restrict__ out) {
  int i = blockIdx.x * blockDim.x + threadIdx.x;
  if (i >= 8 * DM) return;
  int d = i & (DM - 1), b = i >> 9;
  out[i] = h[(((size_t)(b << 8) + 255) << 9) + d];
}

// ---------------------------------------------------------------------------
// Host-side orchestration
// ---------------------------------------------------------------------------
extern "C" void kernel_launch(void* const* d_in, const int* in_sizes, int n_in,
                              void* d_out, int out_size, void* d_ws, size_t ws_size,
                              hipStream_t stream) {
  (void)in_sizes; (void)n_in; (void)out_size; (void)ws_size;
  const float* x    = (const float*)d_in[0];
  const float* ev   = (const float*)d_in[1];
  const float* Wq   = (const float*)d_in[2];  const float* bq  = (const float*)d_in[3];
  const float* Wk   = (const float*)d_in[4];  const float* bk  = (const float*)d_in[5];
  const float* Wv   = (const float*)d_in[6];  const float* bv  = (const float*)d_in[7];
  const float* hQw  = (const float*)d_in[8];  const float* hQb = (const float*)d_in[9];
  const float* hKw  = (const float*)d_in[10]; const float* hKb = (const float*)d_in[11];
  const float* hVw  = (const float*)d_in[12]; const float* hVb = (const float*)d_in[13];
  const float* Wo   = (const float*)d_in[14]; const float* bo  = (const float*)d_in[15];
  const float* ln1g = (const float*)d_in[16]; const float* ln1b= (const float*)d_in[17];
  const float* W1   = (const float*)d_in[18]; const float* b1  = (const float*)d_in[19];
  const float* W2   = (const float*)d_in[20]; const float* b2  = (const float*)d_in[21];
  const float* ln2g = (const float*)d_in[22]; const float* ln2b= (const float*)d_in[23];

  char* ws = (char*)d_ws;
  size_t off = 0;
  auto alloc = [&](size_t bytes) -> void* {
    void* p = (void*)(ws + off);
    off += (bytes + 255) & ~(size_t)255;
    return p;
  };

  const int N = NTOK, D = DM, DK = DKV, H = NH, F = FH, L = NL;
  float*     hbuf = (float*)    alloc((size_t)N * D * 4);
  _Float16*  hf   = (_Float16*) alloc((size_t)N * D * 2);
  _Float16*  q16  = (_Float16*) alloc((size_t)N * DK * 2);
  _Float16*  k16  = (_Float16*) alloc((size_t)N * DK * 2);
  _Float16*  v16  = (_Float16*) alloc((size_t)N * DK * 2);
  _Float16*  Qb   = (_Float16*) alloc((size_t)H * N * DK * 2);
  _Float16*  Kb   = (_Float16*) alloc((size_t)H * N * DK * 2);
  _Float16*  Vb   = (_Float16*) alloc((size_t)H * N * DK * 2);
  _Float16*  Vtb  = (_Float16*) alloc((size_t)H * N * DK * 2);
  _Float16*  cc   = (_Float16*) alloc((size_t)N * D * 2);
  float*     attn = (float*)    alloc((size_t)N * D * 4);   // reused as ffn out
  float*     x1   = (float*)    alloc((size_t)N * D * 4);
  _Float16*  x1h  = (_Float16*) alloc((size_t)N * D * 2);
  _Float16*  y1   = (_Float16*) alloc((size_t)N * F * 2);
  _Float16*  Wqt  = (_Float16*) alloc((size_t)L * DK * D * 2);
  _Float16*  Wkt  = (_Float16*) alloc((size_t)L * DK * D * 2);
  _Float16*  Wvt  = (_Float16*) alloc((size_t)L * DK * D * 2);
  _Float16*  hQwt = (_Float16*) alloc((size_t)L * H * DK * DK * 2);
  _Float16*  hKwt = (_Float16*) alloc((size_t)L * H * DK * DK * 2);
  _Float16*  hVwt = (_Float16*) alloc((size_t)L * H * DK * DK * 2);
  _Float16*  Wot  = (_Float16*) alloc((size_t)L * D * D * 2);
  _Float16*  W1t  = (_Float16*) alloc((size_t)L * F * D * 2);
  _Float16*  W2t  = (_Float16*) alloc((size_t)L * D * F * 2);

  auto prepw = [&](const float* w, _Float16* wt, int K, int Nc) {
    const int tot = K * Nc;
    k_prep_wt<<<(tot + 255) / 256, 256, 0, stream>>>(w, wt, K, Nc);
  };
  // Transpose+convert all weights once (deterministic each call)
  for (int l = 0; l < L; ++l) {
    prepw(Wq + (size_t)l * D * DK, Wqt + (size_t)l * DK * D, D, DK);
    prepw(Wk + (size_t)l * D * DK, Wkt + (size_t)l * DK * D, D, DK);
    prepw(Wv + (size_t)l * D * DK, Wvt + (size_t)l * DK * D, D, DK);
    for (int hh = 0; hh < H; ++hh) {
      const size_t o = ((size_t)l * H + hh) * DK * DK;
      prepw(hQw + o, hQwt + o, DK, DK);
      prepw(hKw + o, hKwt + o, DK, DK);
      prepw(hVw + o, hVwt + o, DK, DK);
    }
    prepw(Wo + (size_t)l * D * D, Wot + (size_t)l * D * D, D, D);
    prepw(W1 + (size_t)l * D * F, W1t + (size_t)l * F * D, D, F);
    prepw(W2 + (size_t)l * F * D, W2t + (size_t)l * D * F, F, D);
  }

  k_prep_h0<<<(N * D) / 256, 256, 0, stream>>>(x, ev, hbuf, hf);

  auto gemm = [&](const _Float16* A, const _Float16* Bt, const float* bias,
                  float* Cf, _Float16* Ch, int M, int K, int Nc, int relu,
                  int batch, long sB, long sBias, long sC) {
    const int jobs = batch * (M / 16) * (Nc / 32);
    k_gemm<<<(jobs + 7) / 8, 256, 0, stream>>>(A, Bt, bias, Cf, Ch, M, K, Nc,
                                               relu, batch, sB, sBias, sC);
  };

  for (int l = 0; l < L; ++l) {
    // base projections q,k,v : [N,512] @ [512,64]
    gemm(hf, Wqt + (size_t)l * DK * D, bq + (size_t)l * DK, nullptr, q16,
         N, D, DK, 0, 1, 0, 0, 0);
    gemm(hf, Wkt + (size_t)l * DK * D, bk + (size_t)l * DK, nullptr, k16,
         N, D, DK, 0, 1, 0, 0, 0);
    gemm(hf, Wvt + (size_t)l * DK * D, bv + (size_t)l * DK, nullptr, v16,
         N, D, DK, 0, 1, 0, 0, 0);
    // per-head projections, batched over the 8 heads : [N,64] @ [64,64]
    gemm(q16, hQwt + (size_t)l * H * DK * DK, hQb + (size_t)l * H * DK,
         nullptr, Qb, N, DK, DK, 0, H, (long)DK * DK, DK, (long)N * DK);
    gemm(k16, hKwt + (size_t)l * H * DK * DK, hKb + (size_t)l * H * DK,
         nullptr, Kb, N, DK, DK, 0, H, (long)DK * DK, DK, (long)N * DK);
    gemm(v16, hVwt + (size_t)l * H * DK * DK, hVb + (size_t)l * H * DK,
         nullptr, Vb, N, DK, DK, 0, H, (long)DK * DK, DK, (long)N * DK);
    k_vt<<<(H * N * DK) / 256, 256, 0, stream>>>(Vb, Vtb);
    // flash attention over the flattened 2048-token axis -> concat [N, H*64] f16
    k_flash<<<(H * (N / 16)) / 8, 256, 0, stream>>>(Qb, Kb, Vtb, cc);
    // output projection + AddNorm1
    gemm(cc, Wot + (size_t)l * D * D, bo + (size_t)l * D, attn, nullptr,
         N, D, D, 0, 1, 0, 0, 0);
    k_addnorm<<<N / 8, 256, 0, stream>>>(attn, hbuf, ln1g + (size_t)l * D,
                                         ln1b + (size_t)l * D, x1, x1h, N);
    // FFN + AddNorm2 (attn buffer reused for ffn output)
    gemm(x1h, W1t + (size_t)l * F * D, b1 + (size_t)l * F, nullptr, y1,
         N, D, F, 1, 1, 0, 0, 0);
    gemm(y1, W2t + (size_t)l * D * F, b2 + (size_t)l * D, attn, nullptr,
         N, F, D, 0, 1, 0, 0, 0);
    k_addnorm<<<N / 8, 256, 0, stream>>>(attn, x1, ln2g + (size_t)l * D,
                                         ln2b + (size_t)l * D, hbuf, hf, N);
  }
  k_extract<<<(8 * DM) / 256, 256, 0, stream>>>(hbuf, (float*)d_out);
}
